// Attention_59708635349115
// MI455X (gfx1250) — matrix operand order repaired and checked by reading, not compiled, not run on hardware
//
#include <hip/hip_runtime.h>

// ---------------- problem constants ----------------
constexpr int CB   = 2;
constexpr int CS   = 2048;
constexpr int CH   = 2048;
constexpr int CNH  = 16;
constexpr int CNKV = 4;
constexpr int CHD  = 128;
constexpr int CNREP = CNH / CNKV;

typedef __attribute__((ext_vector_type(16))) _Float16 v16h;
typedef __attribute__((ext_vector_type(8)))  float    v8f;

union F16Frag {
    v16h  v;
    uint4 q[2];
};

__device__ __forceinline__ v8f wmma16(v16h a, v16h b, v8f c) {
    // D = A(16x32 f16) * B(32x16 f16) + C(16x16 f32)
    return __builtin_amdgcn_wmma_f32_16x16x32_f16(
        /*neg_a=*/false, a, /*neg_b=*/false, b,
        /*c_mod=*/(short)0, c, /*reuse_a=*/false, /*reuse_b=*/false);
}

__device__ __forceinline__ v8f v8f_zero() {
    v8f z;
#pragma unroll
    for (int i = 0; i < 8; ++i) z[i] = 0.0f;
    return z;
}

// B fragment: 8 halves at +0 and 8 halves at +16 (lane = column, per-half-wave K split)
__device__ __forceinline__ v16h load_bfrag(const _Float16* p) {
    F16Frag f;
    f.q[0] = *(const uint4*)(p);
    f.q[1] = *(const uint4*)(p + 16);
    return f.v;
}
// A fragment: 8 halves at +0 and 8 halves at +16 (per-half-wave K split)
__device__ __forceinline__ v16h load_afrag(const _Float16* p) {
    F16Frag f;
    f.q[0] = *(const uint4*)(p);
    f.q[1] = *(const uint4*)(p + 16);
    return f.v;
}

__device__ __forceinline__ float redmax16(float v) {
#pragma unroll
    for (int off = 1; off < 16; off <<= 1)
        v = fmaxf(v, __shfl_xor(v, off, 32));
    return v;
}
__device__ __forceinline__ float redsum16(float v) {
#pragma unroll
    for (int off = 1; off < 16; off <<= 1)
        v += __shfl_xor(v, off, 32);
    return v;
}

// ---------------- fp32 -> fp16 convert (plain) ----------------
__global__ void f32_to_f16_kernel(const float* __restrict__ in,
                                  _Float16* __restrict__ out, int n) {
    int i = blockIdx.x * 256 + threadIdx.x;
    if (i < n) out[i] = (_Float16)in[i];
}

// ---------------- fp32 (K,N) -> fp16 transposed (N,K); N = 2^lgN ----------------
__global__ void cvt_transpose_kernel(const float* __restrict__ in,
                                     _Float16* __restrict__ out, int K, int lgN) {
    int i = blockIdx.x * 256 + threadIdx.x;
    if (i >= (K << lgN)) return;
    int k = i >> lgN;
    int n = i & ((1 << lgN) - 1);          // coalesced read
    out[((size_t)n << 11) + k < 0 ? 0 : (size_t)n * K + k] = (_Float16)in[i];
}

// ---------------- RoPE (in place, head-major f16) ----------------
__global__ void rope_f16_kernel(_Float16* __restrict__ xp,
                                const float* __restrict__ cosT,
                                const float* __restrict__ sinT, int total) {
    int i = blockIdx.x * 256 + threadIdx.x;
    if (i >= total) return;
    int d    = i & 63;
    int srow = i >> 6;            // bh*S + s
    int s    = srow & (CS - 1);
    _Float16* p = xp + (size_t)srow * CHD;
    float x1 = (float)p[d];
    float x2 = (float)p[d + 64];
    float c1 = cosT[s * CHD + d];
    float c2 = cosT[s * CHD + d + 64];
    float s1 = sinT[s * CHD + d];
    float s2 = sinT[s * CHD + d + 64];
    p[d]      = (_Float16)(x1 * c1 - x2 * s1);
    p[d + 64] = (_Float16)(x2 * c2 + x1 * s2);
}

// ---------------- WMMA GEMM (no LDS, transposed weights, SW-pipelined) ----------------
// C[M,N] = A[M,K] * Wt[N,K]^T, f16 in, f32 accumulate.
// Block = 4 waves, block tile 64(M) x 128(N); each wave: 16 rows x 8 N-tiles.
// mode 0: f16 head-major   (((b*n_heads+h)*S + s)*HD + d)
// mode 1: f16 head-major-T (((b*n_heads+h)*HD + d)*S + s)     [for V^T]
// mode 2: f32 row-major to outF
__global__ __launch_bounds__(128) void gemm_wmma_kernel(
    const _Float16* __restrict__ A, const _Float16* __restrict__ Wt,
    _Float16* __restrict__ outH, float* __restrict__ outF,
    int M, int N, int K, int mode, int n_heads) {
    const int tid  = threadIdx.x;
    const int lane = tid & 31;
    const int w    = tid >> 5;          // wave id 0..3
    const int lh   = lane & 15;
    const int hi   = lane >> 4;
    const int n0   = blockIdx.x * 128;
    const int m0   = blockIdx.y * 64;

    v8f c[8];
#pragma unroll
    for (int t = 0; t < 8; ++t) c[t] = v8f_zero();

    const _Float16* arow  = A + (size_t)(m0 + 16 * w + lh) * K + 8 * hi;
    const _Float16* bcol0 = Wt + (size_t)(n0 + lh) * K + 8 * hi;

    // prologue: A and B group 0 (tiles 0..3) at kk=0
    v16h a_cur = load_afrag(arow);
    v16h b0[4];
#pragma unroll
    for (int t = 0; t < 4; ++t)
        b0[t] = load_bfrag(bcol0 + (size_t)(16 * t) * K);

    for (int kk = 0; kk < K; kk += 32) {
        const int kn = (kk + 32 < K) ? kk + 32 : kk;   // clamp (redundant tail reload)
        if (kk + 64 < K) {
            __builtin_prefetch(arow + kk + 64, 0, 1);
            __builtin_prefetch(bcol0 + kk + 64, 0, 1);
        }
        // issue group 1 loads @ kk, then consume group 0
        v16h b1[4];
#pragma unroll
        for (int t = 0; t < 4; ++t)
            b1[t] = load_bfrag(bcol0 + (size_t)(16 * (t + 4)) * K + kk);
#pragma unroll
        for (int t = 0; t < 4; ++t) c[t] = wmma16(a_cur, b0[t], c[t]);

        // issue next-step A + group 0 loads @ kn, then consume group 1
        v16h a_nxt = load_afrag(arow + kn);
#pragma unroll
        for (int t = 0; t < 4; ++t)
            b0[t] = load_bfrag(bcol0 + (size_t)(16 * t) * K + kn);
#pragma unroll
        for (int t = 0; t < 4; ++t) c[t + 4] = wmma16(a_cur, b1[t], c[t + 4]);
        a_cur = a_nxt;
    }

    // epilogue: C layout — lane holds column (lane&15), VGPR r = row r (+8 upper half)
#pragma unroll
    for (int t = 0; t < 8; ++t) {
#pragma unroll
        for (int r = 0; r < 8; ++r) {
            int grow = m0 + 16 * w + r + 8 * hi;
            int gcol = n0 + 16 * t + lh;
            float v  = c[t][r];
            if (mode == 0) {
                int bb = grow >> 11, ss = grow & (CS - 1);
                int hh = gcol >> 7,  dd = gcol & (CHD - 1);
                outH[(((size_t)bb * n_heads + hh) * CS + ss) * CHD + dd] =
                    (_Float16)v;
            } else if (mode == 1) {
                int bb = grow >> 11, ss = grow & (CS - 1);
                int hh = gcol >> 7,  dd = gcol & (CHD - 1);
                outH[(((size_t)bb * n_heads + hh) * CHD + dd) * CS + ss] =
                    (_Float16)v;
            } else {
                outF[(size_t)grow * N + gcol] = v;
            }
        }
    }
}

// ---------------- flash attention (one wave = 16 query rows) ----------------
__global__ __launch_bounds__(32) void flash_attn_kernel(
    const _Float16* __restrict__ Qh,   // (B, NH, S, HD)  f16, RoPE'd
    const _Float16* __restrict__ Kh,   // (B, NKV, S, HD) f16, RoPE'd
    const _Float16* __restrict__ Vt,   // (B, NKV, HD, S) f16 (transposed)
    const float* __restrict__ amask,   // (B, S) f32
    _Float16* __restrict__ Otok) {     // (B*S, NH*HD) f16 token-major
    __shared__ __align__(16) float p_lds[16 * 32];

    const int lane = threadIdx.x;
    const int lh   = lane & 15;
    const int hi   = lane >> 4;
    const int qb   = blockIdx.x;
    const int h    = blockIdx.y;
    const int b    = blockIdx.z;
    const int q0   = qb * 16;
    const int hk   = h / CNREP;

    // resident Q fragments, pre-scaled by 1/sqrt(HD)
    v16h aq[4];
    const _Float16* qrow =
        Qh + (((size_t)b * CNH + h) * CS + q0 + lh) * CHD + 8 * hi;
    const _Float16 qs = (_Float16)0.08838834764831845f;
#pragma unroll
    for (int cc = 0; cc < 4; ++cc) {
        v16h a = load_afrag(qrow + 32 * cc);
        aq[cc] = a * qs;
    }

    v8f o_acc[8];
#pragma unroll
    for (int t = 0; t < 8; ++t) o_acc[t] = v8f_zero();
    float mrow[8], lrow[8];
#pragma unroll
    for (int r = 0; r < 8; ++r) { mrow[r] = -3.0e38f; lrow[r] = 0.0f; }

    const _Float16* Kbase =
        Kh + ((size_t)b * CNKV + hk) * CS * CHD + (size_t)lh * CHD + 8 * hi;
    const _Float16* Vbase =
        Vt + (((size_t)b * CNKV + hk) * CHD + lh) * CS + 8 * hi;

    const int nkb = (qb + 2) >> 1;   // 32-key blocks covering keys 0..q0+15
    for (int it = 0; it < nkb; ++it) {
        const int kb = it * 32;
        const _Float16* k0p = Kbase + (size_t)kb * CHD;

        // ---- issue all V-fragment loads up front (consumed after softmax) ----
        v16h bv[8];
#pragma unroll
        for (int t = 0; t < 8; ++t)
            bv[t] = load_bfrag(Vbase + (size_t)(16 * t) * CS + kb);

        // ---- scores: pipelined K loads, two 16x16 tiles over 32 keys ----
        v8f s0 = v8f_zero(), s1 = v8f_zero();
        v16h bk0 = load_bfrag(k0p);
        v16h bk1 = load_bfrag(k0p + (size_t)16 * CHD);
#pragma unroll
        for (int cc = 0; cc < 4; ++cc) {
            const int nc = (cc < 3) ? cc + 1 : cc;
            v16h nk0 = load_bfrag(k0p + 32 * nc);
            v16h nk1 = load_bfrag(k0p + (size_t)16 * CHD + 32 * nc);
            s0 = wmma16(aq[cc], bk0, s0);
            s1 = wmma16(aq[cc], bk1, s1);
            bk0 = nk0;
            bk1 = nk1;
        }

        const int col0 = kb + lh;
        const int col1 = kb + 16 + lh;
        const float pad0 = (1.0f - amask[(size_t)b * CS + col0]) * -1e30f;
        const float pad1 = (1.0f - amask[(size_t)b * CS + col1]) * -1e30f;

        // ---- online softmax (row stats via 16-lane shfl reductions) ----
#pragma unroll
        for (int r = 0; r < 8; ++r) {
            int row = q0 + r + 8 * hi;
            float sv0 = s0[r] + pad0 + (col0 > row ? -1e30f : 0.0f);
            float sv1 = s1[r] + pad1 + (col1 > row ? -1e30f : 0.0f);
            float lm   = redmax16(fmaxf(sv0, sv1));
            float mnew = fmaxf(mrow[r], lm);
            float p0 = __expf(sv0 - mnew);
            float p1 = __expf(sv1 - mnew);
            float alpha = __expf(mrow[r] - mnew);
            float rs = redsum16(p0 + p1);
            lrow[r] = lrow[r] * alpha + rs;
            mrow[r] = mnew;
#pragma unroll
            for (int t = 0; t < 8; ++t) o_acc[t][r] *= alpha;
            p_lds[(r + 8 * hi) * 32 + lh]      = p0;
            p_lds[(r + 8 * hi) * 32 + 16 + lh] = p1;
        }
        __syncthreads();

        // ---- P (C layout, f32, LDS) -> A fragment (16x32 f16), b128 reads ----
        v16h pf;
        {
            const float* pp = &p_lds[lh * 32 + 8 * hi];
            float4 x0 = *(const float4*)(pp);
            float4 x1 = *(const float4*)(pp + 4);
            float4 y0 = *(const float4*)(pp + 16);
            float4 y1 = *(const float4*)(pp + 20);
            pf[0]  = (_Float16)x0.x; pf[1]  = (_Float16)x0.y;
            pf[2]  = (_Float16)x0.z; pf[3]  = (_Float16)x0.w;
            pf[4]  = (_Float16)x1.x; pf[5]  = (_Float16)x1.y;
            pf[6]  = (_Float16)x1.z; pf[7]  = (_Float16)x1.w;
            pf[8]  = (_Float16)y0.x; pf[9]  = (_Float16)y0.y;
            pf[10] = (_Float16)y0.z; pf[11] = (_Float16)y0.w;
            pf[12] = (_Float16)y1.x; pf[13] = (_Float16)y1.y;
            pf[14] = (_Float16)y1.z; pf[15] = (_Float16)y1.w;
        }

        // ---- O += P @ V (V fragments already resident) ----
#pragma unroll
        for (int t = 0; t < 8; ++t)
            o_acc[t] = wmma16(pf, bv[t], o_acc[t]);
        __syncthreads();
    }

    // ---- epilogue: normalize, write token-major f16 ----
#pragma unroll
    for (int r = 0; r < 8; ++r) {
        float inv = 1.0f / lrow[r];
        int row   = q0 + r + 8 * hi;
        size_t obase = ((size_t)b * CS + row) * (size_t)CH + (size_t)h * CHD;
#pragma unroll
        for (int t = 0; t < 8; ++t)
            Otok[obase + 16 * t + lh] = (_Float16)(o_acc[t][r] * inv);
    }
}

// ---------------- host launch ----------------
extern "C" void kernel_launch(void* const* d_in, const int* in_sizes, int n_in,
                              void* d_out, int out_size, void* d_ws,
                              size_t ws_size, hipStream_t stream) {
    (void)in_sizes; (void)n_in; (void)out_size; (void)ws_size;
    const float* x     = (const float*)d_in[0];
    const float* cosT  = (const float*)d_in[1];
    const float* sinT  = (const float*)d_in[2];
    const float* Wq    = (const float*)d_in[3];
    const float* Wk    = (const float*)d_in[4];
    const float* Wv    = (const float*)d_in[5];
    const float* Wo    = (const float*)d_in[6];
    const float* amask = (const float*)d_in[7];
    float* out = (float*)d_out;

    _Float16* ws = (_Float16*)d_ws;
    size_t off = 0;
    _Float16* xh  = ws + off; off += (size_t)CB * CS * CH;         // x f16
    _Float16* wqt = ws + off; off += (size_t)(CNH * CHD) * CH;     // Wq^T (N,K)
    _Float16* wkt = ws + off; off += (size_t)(CNKV * CHD) * CH;    // Wk^T
    _Float16* wvt = ws + off; off += (size_t)(CNKV * CHD) * CH;    // Wv^T
    _Float16* wot = ws + off; off += (size_t)CH * (CNH * CHD);     // Wo^T
    _Float16* qh  = ws + off; off += (size_t)CB * CNH * CS * CHD;  // Q head-major
    _Float16* kh  = ws + off; off += (size_t)CB * CNKV * CS * CHD; // K head-major
    _Float16* vt  = ws + off; off += (size_t)CB * CNKV * CHD * CS; // V transposed
    _Float16* otok = ws + off;                                     // attn out token-major

    {
        int n = CB * CS * CH;
        f32_to_f16_kernel<<<(n + 255) / 256, 256, 0, stream>>>(x, xh, n);
    }
    auto cvtT = [&](const float* src, _Float16* dst, int K, int lgN) {
        int n = K << lgN;
        cvt_transpose_kernel<<<(n + 255) / 256, 256, 0, stream>>>(src, dst, K, lgN);
    };
    cvtT(Wq, wqt, CH, 11);   // N = 2048
    cvtT(Wk, wkt, CH, 9);    // N = 512
    cvtT(Wv, wvt, CH, 9);    // N = 512
    cvtT(Wo, wot, CNH * CHD, 11);

    const int M = CB * CS;   // 4096 tokens
    // Q/K projections -> head-major f16 ; V projection -> transposed layout
    gemm_wmma_kernel<<<dim3((CNH * CHD) / 128, M / 64), 128, 0, stream>>>(
        xh, wqt, qh, nullptr, M, CNH * CHD, CH, 0, CNH);
    gemm_wmma_kernel<<<dim3((CNKV * CHD) / 128, M / 64), 128, 0, stream>>>(
        xh, wkt, kh, nullptr, M, CNKV * CHD, CH, 0, CNKV);
    gemm_wmma_kernel<<<dim3((CNKV * CHD) / 128, M / 64), 128, 0, stream>>>(
        xh, wvt, vt, nullptr, M, CNKV * CHD, CH, 1, CNKV);

    // RoPE on Q and K (in place)
    {
        int totq = CB * CNH * CS * (CHD / 2);
        rope_f16_kernel<<<(totq + 255) / 256, 256, 0, stream>>>(qh, cosT, sinT, totq);
        int totk = CB * CNKV * CS * (CHD / 2);
        rope_f16_kernel<<<(totk + 255) / 256, 256, 0, stream>>>(kh, cosT, sinT, totk);
    }

    // causal GQA flash attention -> token-major f16
    flash_attn_kernel<<<dim3(CS / 16, CNH, CB), 32, 0, stream>>>(
        qh, kh, vt, amask, otok);

    // output projection -> fp32 d_out
    gemm_wmma_kernel<<<dim3(CH / 128, M / 64), 128, 0, stream>>>(
        otok, wot, nullptr, out, M, CH, CNH * CHD, 2, CNH);
}
